// GNNModel_89137751261683
// MI455X (gfx1250) — compile-verified, matrix operand order checked by reading, not run on hardware
//
#include <hip/hip_runtime.h>
#include <hip/hip_bf16.h>

// ---------------------------------------------------------------------------
// N=500000 nodes, K=32 in, H=256 hidden. 4-layer MLP, tanh x3, log_sigmoid.
// f16 WMMA (v_wmma_f32_16x16x32_f16); weights transposed->f16 in ws, staged
// into LDS with bank-conflict padding; biases/W4 as f16 in LDS; hardware
// v_tanh_f32; one 16-row tile per wave, persistent waves.
// LDS layout ordered so act+bias (many small-offset DS ops) sit below the
// 64KB DS-immediate limit: act | bias | W1^T | W2^T | W3^T.
// ---------------------------------------------------------------------------

typedef _Float16 v16h __attribute__((ext_vector_type(16)));
typedef _Float16 h8   __attribute__((ext_vector_type(8)));
typedef float    v8f  __attribute__((ext_vector_type(8)));
typedef float    f4   __attribute__((ext_vector_type(4)));

union H16 { v16h v; h8 h[2]; };

#define N_NODES   500000
#define NUM_TILES (N_NODES / 16)   // 31250, exact

// LDS layout (byte offsets). +8-half row padding on the 256-wide matrices so
// 16 lanes' b128 loads stride 528B -> distinct bank groups (132 % 64 = 4/row).
#define SW1 40                          // W1^T row stride in halves (32 + 8)
#define SW  264                         // W2^T/W3^T/act row stride (256 + 8)
#define ACT_BYTES (16 * SW * 2)         // 8448 per wave
#define OFF_ACT  0                      // 4 waves * 8448 = 33792 (< 64KB imm)
#define OFF_BIAS (4 * ACT_BYTES)        // 33792 .. 35840  (< 64KB imm)
#define OFF_W1   (OFF_BIAS + 4 * 256 * 2)   // 35840
#define OFF_W2   (OFF_W1 + 256 * SW1 * 2)   // 56320
#define OFF_W3   (OFF_W2 + 256 * SW * 2)    // 191488
#define LDS_BYTES (OFF_W3 + 256 * SW * 2)   // 326656 <= 327680 (320KB)

// Branch-free tanh: hardware V_TANH_F32 (probe-confirmed in round-2 asm).
__device__ __forceinline__ float fast_tanh(float x) {
#if __has_builtin(__builtin_amdgcn_tanhf)
  return __builtin_amdgcn_tanhf(x);
#elif __has_builtin(__builtin_amdgcn_tanh_f32)
  return __builtin_amdgcn_tanh_f32(x);
#else
  float e = __expf(-2.0f * __builtin_fabsf(x));
  float t = (1.0f - e) * __builtin_amdgcn_rcpf(1.0f + e);
  return __builtin_copysignf(t, x);
#endif
}

// ---------------------------------------------------------------------------
// Prep: transpose + f32->f16 convert weights into workspace.
// w1t[n*32+k] = W1[k][n], w2t[n*256+k] = W2[k][n], w3t likewise.
// ---------------------------------------------------------------------------
__global__ void prep_weights(const float* __restrict__ W1,
                             const float* __restrict__ W2,
                             const float* __restrict__ W3,
                             _Float16* __restrict__ w1t,
                             _Float16* __restrict__ w2t,
                             _Float16* __restrict__ w3t) {
  int stride = gridDim.x * blockDim.x;
  int tid = blockIdx.x * blockDim.x + threadIdx.x;
  for (int i = tid; i < 256 * 32; i += stride) {
    int n = i >> 5, k = i & 31;
    w1t[i] = (_Float16)W1[k * 256 + n];
  }
  for (int i = tid; i < 256 * 256; i += stride) {
    int n = i >> 8, k = i & 255;
    w2t[i] = (_Float16)W2[k * 256 + n];
    w3t[i] = (_Float16)W3[k * 256 + n];
  }
}

// ---------------------------------------------------------------------------
// Main fused MLP kernel. 128 threads = 4 waves per WG; one 16-row tile/wave.
// ---------------------------------------------------------------------------
__global__ __launch_bounds__(128)
void mlp_fused(const float* __restrict__ edge_attr,
               const _Float16* __restrict__ w1t,
               const _Float16* __restrict__ w2t,
               const _Float16* __restrict__ w3t,
               const float* __restrict__ b1,
               const float* __restrict__ b2,
               const float* __restrict__ b3,
               const float* __restrict__ w4,
               const float* __restrict__ b4,
               float* __restrict__ out) {
  extern __shared__ __align__(16) char smem[];
  _Float16* lw1 = (_Float16*)(smem + OFF_W1);
  _Float16* lw2 = (_Float16*)(smem + OFF_W2);
  _Float16* lw3 = (_Float16*)(smem + OFF_W3);
  _Float16* lbs = (_Float16*)(smem + OFF_BIAS);   // [b1|b2|b3|w4], 256 each

  const int tid = threadIdx.x;

  // ---- stage weights into LDS (padded rows), coalesced b128 chunks ----
  for (int i = tid; i < 256 * 4; i += 128) {            // W1^T: 4 chunks/row
    int n = i >> 2, c = i & 3;
    *(h8*)(lw1 + n * SW1 + c * 8) = *(const h8*)(w1t + n * 32 + c * 8);
  }
  for (int i = tid; i < 256 * 32; i += 128) {           // W2^T/W3^T: 32 chunks/row
    int n = i >> 5, c = i & 31;
    *(h8*)(lw2 + n * SW + c * 8) = *(const h8*)(w2t + i * 8);
    *(h8*)(lw3 + n * SW + c * 8) = *(const h8*)(w3t + i * 8);
  }
  for (int i = tid; i < 256; i += 128) {                // biases + W4 -> f16
    lbs[i]       = (_Float16)b1[i];
    lbs[256 + i] = (_Float16)b2[i];
    lbs[512 + i] = (_Float16)b3[i];
    lbs[768 + i] = (_Float16)w4[i];
  }
  __syncthreads();

  const int wave = tid >> 5;
  const int lane = tid & 31;
  const int ln = lane & 15;   // A: row M; B/C: column N
  const int hi = lane >> 4;   // K-half selector per ISA 16-bit layouts
  _Float16* act = (_Float16*)(smem + OFF_ACT) + wave * (16 * SW);

  const float bias4 = b4[0];
  const int waves_total = gridDim.x * 4;

  for (int tile = blockIdx.x * 4 + wave; tile < NUM_TILES; tile += waves_total) {
    // ================= Layer 1: [16x32] @ W1 =================
    H16 a1;
    {
      const f4* g4 = (const f4*)(edge_attr + (size_t)(tile * 16 + ln) * 32);
      f4 x0 = g4[hi * 2], x1 = g4[hi * 2 + 1];        // K = hi*8 .. hi*8+7
      f4 x2 = g4[4 + hi * 2], x3 = g4[5 + hi * 2];    // K = 16+hi*8 ..
      #pragma unroll
      for (int i = 0; i < 4; i++) {
        a1.h[0][i]     = (_Float16)x0[i];
        a1.h[0][4 + i] = (_Float16)x1[i];
        a1.h[1][i]     = (_Float16)x2[i];
        a1.h[1][4 + i] = (_Float16)x3[i];
      }
    }
    #pragma unroll 4
    for (int nt = 0; nt < 16; nt++) {
      int col = nt * 16 + ln;
      const _Float16* bcol = lw1 + col * SW1;
      H16 b;
      b.h[0] = *(const h8*)(bcol + hi * 8);
      b.h[1] = *(const h8*)(bcol + 16 + hi * 8);
      v8f c = {};
      c = __builtin_amdgcn_wmma_f32_16x16x32_f16(false, a1.v, false, b.v,
                                                 (short)0, c, false, false);
      float bb = (float)lbs[col];
      #pragma unroll
      for (int j = 0; j < 8; j++)
        act[(j + 8 * hi) * SW + col] = (_Float16)fast_tanh(c[j] + bb);
    }
    asm volatile("s_wait_dscnt 0x0" ::: "memory");

    // ================= Layer 2: [16x256] @ W2 =================
    H16 apan[8];
    #pragma unroll
    for (int ks = 0; ks < 8; ks++) {                   // whole A panel -> 64 VGPRs
      const _Float16* ar = act + ln * SW + ks * 32;
      apan[ks].h[0] = *(const h8*)(ar + hi * 8);
      apan[ks].h[1] = *(const h8*)(ar + 16 + hi * 8);
    }
    #pragma unroll 4
    for (int nt = 0; nt < 16; nt++) {
      int col = nt * 16 + ln;
      const _Float16* brow = lw2 + col * SW;
      float bb = (float)lbs[256 + col];
      v8f c = {};
      #pragma unroll
      for (int ks = 0; ks < 8; ks++) {
        H16 b;
        b.h[0] = *(const h8*)(brow + ks * 32 + hi * 8);
        b.h[1] = *(const h8*)(brow + ks * 32 + 16 + hi * 8);
        c = __builtin_amdgcn_wmma_f32_16x16x32_f16(false, apan[ks].v, false, b.v,
                                                   (short)0, c, false, false);
      }
      #pragma unroll
      for (int j = 0; j < 8; j++)
        act[(j + 8 * hi) * SW + col] = (_Float16)fast_tanh(c[j] + bb);
    }
    asm volatile("s_wait_dscnt 0x0" ::: "memory");

    // ========= Layer 3 (+ fused layer-4 partial dot with W4) =========
    #pragma unroll
    for (int ks = 0; ks < 8; ks++) {
      const _Float16* ar = act + ln * SW + ks * 32;
      apan[ks].h[0] = *(const h8*)(ar + hi * 8);
      apan[ks].h[1] = *(const h8*)(ar + 16 + hi * 8);
    }
    float p[8] = {};
    #pragma unroll 4
    for (int nt = 0; nt < 16; nt++) {
      int col = nt * 16 + ln;
      const _Float16* brow = lw3 + col * SW;
      float bb = (float)lbs[512 + col];
      float wc = (float)lbs[768 + col];
      v8f c = {};
      #pragma unroll
      for (int ks = 0; ks < 8; ks++) {
        H16 b;
        b.h[0] = *(const h8*)(brow + ks * 32 + hi * 8);
        b.h[1] = *(const h8*)(brow + ks * 32 + 16 + hi * 8);
        c = __builtin_amdgcn_wmma_f32_16x16x32_f16(false, apan[ks].v, false, b.v,
                                                   (short)0, c, false, false);
      }
      #pragma unroll
      for (int j = 0; j < 8; j++)
        p[j] += fast_tanh(c[j] + bb) * wc;             // row m = j+8*hi
    }

    // ===== Layer 4 reduce across the 16 lanes holding one row's columns =====
    #pragma unroll
    for (int j = 0; j < 8; j++) {
      float r = p[j];
      r += __shfl_xor(r, 1);
      r += __shfl_xor(r, 2);
      r += __shfl_xor(r, 4);
      r += __shfl_xor(r, 8);
      p[j] = r;
    }
    if (ln == 0) {                                     // lanes 0 and 16 write 8 rows each
      #pragma unroll
      for (int j = 0; j < 8; j++) {
        float xv = p[j] + bias4;
        // log_sigmoid(x) = min(x,0) - log(1 + exp(-|x|)), fast-math intrinsics
        out[tile * 16 + j + 8 * hi] =
            __builtin_fminf(xv, 0.0f) - __logf(1.0f + __expf(-__builtin_fabsf(xv)));
      }
    }
  }
}

// ---------------------------------------------------------------------------
extern "C" void kernel_launch(void* const* d_in, const int* in_sizes, int n_in,
                              void* d_out, int out_size, void* d_ws, size_t ws_size,
                              hipStream_t stream) {
  (void)in_sizes; (void)n_in; (void)out_size; (void)ws_size;
  // setup_inputs order: x, edge_index, edge_attr, W1,b1, W2,b2, W3,b3, W4,b4
  const float* edge_attr = (const float*)d_in[2];
  const float* W1 = (const float*)d_in[3];
  const float* b1 = (const float*)d_in[4];
  const float* W2 = (const float*)d_in[5];
  const float* b2 = (const float*)d_in[6];
  const float* W3 = (const float*)d_in[7];
  const float* b3 = (const float*)d_in[8];
  const float* W4 = (const float*)d_in[9];
  const float* b4 = (const float*)d_in[10];
  float* out = (float*)d_out;

  _Float16* w1t = (_Float16*)d_ws;           // 32*256 halves
  _Float16* w2t = w1t + 32 * 256;            // 256*256 halves
  _Float16* w3t = w2t + 256 * 256;           // 256*256 halves

  prep_weights<<<64, 256, 0, stream>>>(W1, W2, W3, w1t, w2t, w3t);

  (void)hipFuncSetAttribute((const void*)mlp_fused,
                            hipFuncAttributeMaxDynamicSharedMemorySize, LDS_BYTES);
  mlp_fused<<<2048, 128, LDS_BYTES, stream>>>(edge_attr, w1t, w2t, w3t,
                                              b1, b2, b3, W4, b4, out);
}
